// GNNStack_87909390615131
// MI455X (gfx1250) — matrix operand order repaired and checked by reading, not compiled, not run on hardware
//
#include <hip/hip_runtime.h>
#include <hip/hip_bf16.h>
#include <stdint.h>

// ---------------- problem constants (from reference) ----------------
#define NNODES 50000
#define NEDGES 600000
#define INCH   128
#define HC     128     // HEADS*OUT_CH
#define EDIM   64
#define OUTC   32
#define NLAYER 3
#define NEG_SLOPE 0.2f
#define BN_EPS 1e-5f

typedef __attribute__((ext_vector_type(16))) _Float16 v16h;
typedef __attribute__((ext_vector_type(8)))  float    v8f;

// ------------- ordered-uint encoding for float atomic max -------------
__device__ __forceinline__ unsigned ordKey(float f) {
  unsigned u = __float_as_uint(f);
  return (u & 0x80000000u) ? ~u : (u | 0x80000000u);
}
__device__ __forceinline__ float ordDecode(unsigned u) {
  return (u & 0x80000000u) ? __uint_as_float(u & 0x7FFFFFFFu) : __uint_as_float(~u);
}

// ---------------------------- fill ----------------------------
__global__ void fill_u32(unsigned* __restrict__ p, unsigned v, size_t n) {
  size_t i = (size_t)blockIdx.x * blockDim.x + threadIdx.x;
  if (i < n) p[i] = v;
}

// ------------- pack A operand (f32 [M,K] -> WMMA f16 A layout) -------------
// layout: dst[((rt*KS + ks)*32 + lane)*16 + j]
//   m = rt*16 + (lane&15)
//   k = ks*32 + (j>>3)*16 + (lane>>4)*8 + (j&7)
__global__ void pack_a_f16(const float* __restrict__ src, _Float16* __restrict__ dst,
                           int Mtiles, int KS, int K) {
  size_t total = (size_t)Mtiles * KS * 512;
  size_t stride = (size_t)gridDim.x * blockDim.x;
  for (size_t p = (size_t)blockIdx.x * blockDim.x + threadIdx.x; p < total; p += stride) {
    int j = (int)(p & 15);
    int l = (int)((p >> 4) & 31);
    size_t t = p >> 9;
    int ks = (int)(t % KS);
    size_t rt = t / KS;
    size_t m = rt * 16 + (l & 15);
    int k = ks * 32 + ((j >> 3) << 4) + ((l >> 4) << 3) + (j & 7);
    dst[p] = (_Float16)src[m * (size_t)K + k];
  }
}

// ------------- pack B operand (f32 [K,N=128] -> WMMA f16 B layout) -------------
// dst[((ct*KS + ks)*32 + lane)*16 + j]
//   k = ks*32 + (j>>3)*16 + (lane>>4)*8 + (j&7) ; col = ct*16 + (lane&15)
__global__ void pack_b_f16(const float* __restrict__ src, _Float16* __restrict__ dst,
                           int KS, int N) {
  size_t total = (size_t)(N / 16) * KS * 512;
  size_t stride = (size_t)gridDim.x * blockDim.x;
  for (size_t p = (size_t)blockIdx.x * blockDim.x + threadIdx.x; p < total; p += stride) {
    int j = (int)(p & 15);
    int l = (int)((p >> 4) & 31);
    size_t t = p >> 9;
    int ks = (int)(t % KS);
    int ct = (int)(t / KS);
    int k = ks * 32 + ((j >> 3) << 4) + ((l >> 4) << 3) + (j & 7);
    int col = ct * 16 + (l & 15);
    dst[p] = (_Float16)src[(size_t)k * N + col];
  }
}

// ------------- WMMA GEMM: C[M,128] = Apack x Bpack (f16 in, f32 out) -------------
// block = 256 threads = 8 waves; wave w handles column tile w; blockIdx.x = row tile.
// optional per-column scale (used to fold att_e into the edge GEMM epilogue).
__global__ void wmma_gemm_f16(const _Float16* __restrict__ Ap, const _Float16* __restrict__ Bp,
                              float* __restrict__ C, int Mtiles, int KS,
                              const float* __restrict__ colScale) {
  const int N = 128;
  int wave = threadIdx.x >> 5;       // 0..7 -> column tile
  int lane = threadIdx.x & 31;
  int rt = blockIdx.x;
  if (rt >= Mtiles) return;
  const _Float16* a = Ap + ((size_t)rt * KS) * 512 + lane * 16;
  const _Float16* b = Bp + ((size_t)wave * KS) * 512 + lane * 16;
  v8f acc = {};
  for (int ks = 0; ks < KS; ++ks) {
    v16h av = *(const v16h*)(a + (size_t)ks * 512);
    v16h bv = *(const v16h*)(b + (size_t)ks * 512);
    acc = __builtin_amdgcn_wmma_f32_16x16x32_f16(false, av, false, bv,
                                                 (short)0, acc, false, false);
  }
  int n = wave * 16 + (lane & 15);
  float s = colScale ? colScale[n] : 1.0f;
  int m0 = rt * 16 + ((lane >> 4) << 3);
#pragma unroll
  for (int r = 0; r < 8; ++r)
    C[(size_t)(m0 + r) * N + n] = acc[r] * s;
}

// ------------- edge pass 1: alpha = lrelu(ar[dst]+al[src]+ae); segment max -------------
__global__ void edge_alpha_max(const float* __restrict__ xt, float* __restrict__ ae,
                               const long long* __restrict__ esrc, const long long* __restrict__ edst,
                               const float* __restrict__ attl, const float* __restrict__ attr,
                               unsigned* __restrict__ mbuf, size_t total) {
  size_t i = (size_t)blockIdx.x * blockDim.x + threadIdx.x;
  if (i >= total) return;
  size_t e = i >> 7;
  int c = (int)(i & 127);
  size_t s = (size_t)esrc[e];
  size_t d = (size_t)edst[e];
  float a = attr[c] * xt[d * HC + c] + attl[c] * xt[s * HC + c] + ae[i];
  a = (a > 0.0f) ? a : NEG_SLOPE * a;
  ae[i] = a;
  atomicMax(&mbuf[d * HC + c], ordKey(a));
}

// ------------- edge pass 2: ex = exp(alpha - m[dst]); segment sum -------------
__global__ void edge_exp_sum(float* __restrict__ ae, const unsigned* __restrict__ mbuf,
                             float* __restrict__ denom, const long long* __restrict__ edst,
                             size_t total) {
  size_t i = (size_t)blockIdx.x * blockDim.x + threadIdx.x;
  if (i >= total) return;
  size_t e = i >> 7;
  int c = (int)(i & 127);
  size_t d = (size_t)edst[e];
  float ex = expf(ae[i] - ordDecode(mbuf[d * HC + c]));
  ae[i] = ex;
  atomicAdd(&denom[d * HC + c], ex);
}

// ------------- edge pass 3: out[dst] += ex/(denom+eps) * xt[src] -------------
__global__ void edge_aggregate(const float* __restrict__ ae, const float* __restrict__ denom,
                               const float* __restrict__ xt, float* __restrict__ out,
                               const long long* __restrict__ esrc, const long long* __restrict__ edst,
                               size_t total) {
  size_t i = (size_t)blockIdx.x * blockDim.x + threadIdx.x;
  if (i >= total) return;
  size_t e = i >> 7;
  int c = (int)(i & 127);
  size_t s = (size_t)esrc[e];
  size_t d = (size_t)edst[e];
  float attn = ae[i] / (denom[d * HC + c] + 1e-16f);
  atomicAdd(&out[d * HC + c], attn * xt[s * HC + c]);
}

// ------------- BN: per-channel sum / sumsq reduction -------------
__global__ void bn_reduce(const float* __restrict__ h, float* __restrict__ stats, size_t total) {
  __shared__ float ls[256];
  ls[threadIdx.x] = 0.0f;
  __syncthreads();
  size_t stride = (size_t)gridDim.x * blockDim.x;
  for (size_t i = (size_t)blockIdx.x * blockDim.x + threadIdx.x; i < total; i += stride) {
    int c = (int)(i & 127);
    float v = h[i];
    atomicAdd(&ls[c], v);
    atomicAdd(&ls[128 + c], v * v);
  }
  __syncthreads();
  atomicAdd(&stats[threadIdx.x], ls[threadIdx.x]);
}

// ------------- BN apply + ReLU, fused with next-layer A-operand repack -------------
__global__ void bn_apply_pack(const float* __restrict__ h, const float* __restrict__ stats,
                              const float* __restrict__ gamma, const float* __restrict__ beta,
                              _Float16* __restrict__ dst, int Mtiles) {
  const int KS = HC / 32, K = HC;
  size_t total = (size_t)Mtiles * KS * 512;
  size_t stride = (size_t)gridDim.x * blockDim.x;
  const float invN = 1.0f / (float)NNODES;
  for (size_t p = (size_t)blockIdx.x * blockDim.x + threadIdx.x; p < total; p += stride) {
    int j = (int)(p & 15);
    int l = (int)((p >> 4) & 31);
    size_t t = p >> 9;
    int ks = (int)(t % KS);
    size_t rt = t / KS;
    size_t m = rt * 16 + (l & 15);
    int k = ks * 32 + ((j >> 3) << 4) + ((l >> 4) << 3) + (j & 7);
    float mean = stats[k] * invN;
    float var = stats[128 + k] * invN - mean * mean;
    float v = h[m * (size_t)K + k];
    v = gamma[k] * (v - mean) * rsqrtf(var + BN_EPS) + beta[k];
    v = (v > 0.0f) ? v : 0.0f;
    dst[p] = (_Float16)v;
  }
}

// ------------- final MLP: 128 -> 32 -> 32 -------------
__global__ void mlp1(const float* __restrict__ h, const float* __restrict__ W1,
                     const float* __restrict__ b1, float* __restrict__ tmp) {
  size_t i = (size_t)blockIdx.x * blockDim.x + threadIdx.x;
  size_t total = (size_t)NNODES * OUTC;
  if (i >= total) return;
  size_t n = i >> 5;
  int j = (int)(i & 31);
  const float* hr = h + n * HC;
  float s = b1[j];
#pragma unroll 8
  for (int k = 0; k < HC; ++k) s += hr[k] * W1[k * OUTC + j];
  tmp[i] = s;
}
__global__ void mlp2(const float* __restrict__ t, const float* __restrict__ W2,
                     const float* __restrict__ b2, float* __restrict__ out) {
  size_t i = (size_t)blockIdx.x * blockDim.x + threadIdx.x;
  size_t total = (size_t)NNODES * OUTC;
  if (i >= total) return;
  size_t n = i >> 5;
  int j = (int)(i & 31);
  const float* tr = t + n * OUTC;
  float s = b2[j];
#pragma unroll
  for (int k = 0; k < OUTC; ++k) s += tr[k] * W2[k * OUTC + j];
  out[i] = s;
}

// ---------------------------------------------------------------------------
static inline char* alignup(char* p, size_t a) {
  return (char*)(((uintptr_t)p + a - 1) & ~(uintptr_t)(a - 1));
}

extern "C" void kernel_launch(void* const* d_in, const int* in_sizes, int n_in,
                              void* d_out, int out_size, void* d_ws, size_t ws_size,
                              hipStream_t stream) {
  const float*     x        = (const float*)d_in[0];
  const long long* eidx     = (const long long*)d_in[1];   // int64 in reference
  const float*     eattr    = (const float*)d_in[2];
  const float*     W_l      = (const float*)d_in[3];       // [3,128,128]
  const float*     att_l    = (const float*)d_in[4];       // [3,128]
  const float*     att_r    = (const float*)d_in[5];
  const float*     W_e      = (const float*)d_in[6];       // [3,64,128]
  const float*     att_e    = (const float*)d_in[7];
  const float*     bn_gamma = (const float*)d_in[8];       // [2,128]
  const float*     bn_beta  = (const float*)d_in[9];
  const float*     W1       = (const float*)d_in[10];
  const float*     b1       = (const float*)d_in[11];
  const float*     W2       = (const float*)d_in[12];
  const float*     b2       = (const float*)d_in[13];
  (void)in_sizes; (void)n_in; (void)out_size; (void)ws_size;

  const long long* esrc = eidx;
  const long long* edst = eidx + NEDGES;

  // ---- carve workspace ----
  char* w = (char*)d_ws;
  float*    xt    = (float*)w;                 w = alignup(w + (size_t)NNODES * HC * 4, 256);
  float*    hout  = (float*)w;                 w = alignup(w + (size_t)NNODES * HC * 4, 256);
  unsigned* mbuf  = (unsigned*)w;              w = alignup(w + (size_t)NNODES * HC * 4, 256);
  float*    denom = (float*)w;                 w = alignup(w + (size_t)NNODES * HC * 4, 256);
  float*    ae    = (float*)w;                 w = alignup(w + (size_t)NEDGES * HC * 4, 256);
  _Float16* ApN   = (_Float16*)w;              w = alignup(w + (size_t)NNODES * HC * 2, 256);
  _Float16* ApE   = (_Float16*)w;              w = alignup(w + (size_t)NEDGES * EDIM * 2, 256);
  _Float16* BpW   = (_Float16*)w;              w = alignup(w + (size_t)HC * HC * 2, 256);
  _Float16* BpWe  = (_Float16*)w;              w = alignup(w + (size_t)EDIM * HC * 2, 256);
  float*    stats = (float*)w;                 w = alignup(w + 256 * 4, 256);
  float*    tmp32 = (float*)w;                 w = alignup(w + (size_t)NNODES * OUTC * 4, 256);

  const int B = 256;
  const int nodeMtiles = NNODES / 16;          // 3125
  const int edgeMtiles = NEDGES / 16;          // 37500
  const size_t nodeElems = (size_t)NNODES * HC;    // 6.4M
  const size_t edgeElems = (size_t)NEDGES * HC;    // 76.8M
  const size_t packN = (size_t)nodeMtiles * 4 * 512;
  const size_t packE = (size_t)edgeMtiles * 2 * 512;

  // layer-0 A operand: pack x (f32 -> f16 WMMA layout)
  pack_a_f16<<<(unsigned)((packN + B - 1) / B), B, 0, stream>>>(x, ApN, nodeMtiles, 4, INCH);
  // edge A operand: constant across layers
  pack_a_f16<<<(unsigned)((packE + B - 1) / B), B, 0, stream>>>(eattr, ApE, edgeMtiles, 2, EDIM);

  for (int i = 0; i < NLAYER; ++i) {
    // pack per-layer weights into WMMA B layout
    pack_b_f16<<<(unsigned)(((size_t)8 * 4 * 512 + B - 1) / B), B, 0, stream>>>(
        W_l + (size_t)i * INCH * HC, BpW, 4, HC);
    pack_b_f16<<<(unsigned)(((size_t)8 * 2 * 512 + B - 1) / B), B, 0, stream>>>(
        W_e + (size_t)i * EDIM * HC, BpWe, 2, HC);

    // xt = h @ W   (WMMA)
    wmma_gemm_f16<<<nodeMtiles, B, 0, stream>>>(ApN, BpW, xt, nodeMtiles, 4, nullptr);
    // ae = (edge_attr @ W_e) * att_e   (WMMA, att_e folded into epilogue)
    wmma_gemm_f16<<<edgeMtiles, B, 0, stream>>>(ApE, BpWe, ae, edgeMtiles, 2,
                                                att_e + (size_t)i * HC);

    // init segment buffers
    fill_u32<<<(unsigned)((nodeElems + B - 1) / B), B, 0, stream>>>(mbuf, 0u, nodeElems);
    fill_u32<<<(unsigned)((nodeElems + B - 1) / B), B, 0, stream>>>((unsigned*)denom, 0u, nodeElems);
    fill_u32<<<(unsigned)((nodeElems + B - 1) / B), B, 0, stream>>>((unsigned*)hout, 0u, nodeElems);

    // three streaming edge passes (alpha stored in-place in ae)
    unsigned egrid = (unsigned)((edgeElems + B - 1) / B);
    edge_alpha_max<<<egrid, B, 0, stream>>>(xt, ae, esrc, edst,
                                            att_l + (size_t)i * HC, att_r + (size_t)i * HC,
                                            mbuf, edgeElems);
    edge_exp_sum<<<egrid, B, 0, stream>>>(ae, mbuf, denom, edst, edgeElems);
    edge_aggregate<<<egrid, B, 0, stream>>>(ae, denom, xt, hout, esrc, edst, edgeElems);

    if (i != NLAYER - 1) {
      fill_u32<<<1, 256, 0, stream>>>((unsigned*)stats, 0u, 256);
      bn_reduce<<<512, B, 0, stream>>>(hout, stats, nodeElems);
      // BN + ReLU fused with repack into next layer's A operand
      bn_apply_pack<<<(unsigned)((packN + B - 1) / B), B, 0, stream>>>(
          hout, stats, bn_gamma + (size_t)i * HC, bn_beta + (size_t)i * HC, ApN, nodeMtiles);
    }
  }

  // final MLP: h @ W1 + b1, then @ W2 + b2 -> d_out [N,32] f32
  size_t mt = (size_t)NNODES * OUTC;
  mlp1<<<(unsigned)((mt + B - 1) / B), B, 0, stream>>>(hout, W1, b1, tmp32);
  mlp2<<<(unsigned)((mt + B - 1) / B), B, 0, stream>>>(tmp32, W2, b2, (float*)d_out);
}
